// FreeEnergyMixer_57389353009777
// MI455X (gfx1250) — matrix operand order repaired and checked by reading, not compiled, not run on hardware
//
#include <hip/hip_runtime.h>
#include <hip/hip_bf16.h>

// ---------------------------------------------------------------------------
// Gated linear attention read, chunk-parallel WMMA formulation for gfx1250.
//
// Recurrence (exactly equals the reference incl. clip):
//   cg_t = clip(cumsum(a), -60, 50)      (monotone non-increasing)
//   S_t  = exp(cg_t - cg_{t-1}) S_{t-1} + k_t v_t^T
//   Ks_t = exp(cg_t - cg_{t-1}) Ks_{t-1} + k_t
//   out_t = (q_t . S_t) / (q_t . Ks_t + 1e-5)
// Chunked (C=32): intra-chunk via v_wmma_f32_16x16x32_f16, state carried in
// LDS (f32). Grid = B*H*ESPL (e split 4-ways) = 64 blocks x 128 threads.
// ---------------------------------------------------------------------------

typedef __attribute__((ext_vector_type(16))) _Float16 v16h;
typedef __attribute__((ext_vector_type(8)))  float    v8f;

namespace {
constexpr int  Bb   = 2;
constexpr int  T    = 1024;
constexpr int  H    = 8;
constexpr int  Dh   = 64;    // q/k head dim
constexpr int  Ev   = 64;    // v head dim
constexpr int  Dx   = 512;   // x feature dim
constexpr int  C    = 32;    // chunk length
constexpr int  NC   = T / C;
constexpr int  ESPL = 4;     // e-dim split across blocks
constexpr int  ESL  = Ev / ESPL;   // 16
constexpr int  QSTR = Dh + 4;      // 68 (272B row, 16B aligned)
constexpr int  VSTR = ESL + 4;     // 20 (80B row, 16B aligned)
constexpr int  ASTR = C + 4;       // 36 (144B row, 16B aligned)
constexpr int  SSTR = ESL + 4;     // 20
constexpr float kEps = 1e-6f;
}

// ---- WMMA operand loaders (f32 LDS -> f16 register layout per ISA spec) ----

// A-matrix 16x32 f16: lane&15 = M, lane>>4 selects K-half; K(v)=2(v&3)+8*half+16*(v>>2)
__device__ __forceinline__ v16h ldA(const float* s, int stride, int m0, int k0) {
  const int lane = threadIdx.x & 31;
  const int M = lane & 15, half = lane >> 4;
  const float2* row2 = (const float2*)(s + (m0 + M) * stride + k0);  // even offsets
  v16h r;
#pragma unroll
  for (int v = 0; v < 8; ++v) {
    const int K = ((v & 3) << 1) + (half << 3) + ((v >> 2) << 4);
    float2 p = row2[K >> 1];
    r[2 * v]     = (_Float16)p.x;
    r[2 * v + 1] = (_Float16)p.y;
  }
  return r;
}

// A-matrix with per-row scale (used for exp(r_i) * q_i)
__device__ __forceinline__ v16h ldA_rowscale(const float* s, int stride, int m0,
                                             int k0, const float* wrow) {
  const int lane = threadIdx.x & 31;
  const int M = lane & 15, half = lane >> 4;
  const float  sc    = wrow[m0 + M];
  const float2* row2 = (const float2*)(s + (m0 + M) * stride + k0);
  v16h r;
#pragma unroll
  for (int v = 0; v < 8; ++v) {
    const int K = ((v & 3) << 1) + (half << 3) + ((v >> 2) << 4);
    float2 p = row2[K >> 1];
    r[2 * v]     = (_Float16)(p.x * sc);
    r[2 * v + 1] = (_Float16)(p.y * sc);
  }
  return r;
}

// Transposed A with per-K scale: element(M,K) = s[(k0+K)*stride + m0+M] * wk[k0+K]
__device__ __forceinline__ v16h ldAT_kscale(const float* s, int stride, int m0,
                                            int k0, const float* wk) {
  const int lane = threadIdx.x & 31;
  const int M = lane & 15, half = lane >> 4;
  v16h r;
#pragma unroll
  for (int v = 0; v < 8; ++v) {
    const int K = ((v & 3) << 1) + (half << 3) + ((v >> 2) << 4) + k0;
    r[2 * v]     = (_Float16)(s[K * stride + m0 + M]       * wk[K]);
    r[2 * v + 1] = (_Float16)(s[(K + 1) * stride + m0 + M] * wk[K + 1]);
  }
  return r;
}

// B-matrix 32x16 f16: lane&15 = N, K(v)=2(v&3)+8*half+16*(v>>2)
__device__ __forceinline__ v16h ldB(const float* s, int stride, int k0, int n0) {
  const int lane = threadIdx.x & 31;
  const int N = lane & 15, half = lane >> 4;
  v16h r;
#pragma unroll
  for (int v = 0; v < 8; ++v) {
    const int K = ((v & 3) << 1) + (half << 3) + ((v >> 2) << 4) + k0;
    r[2 * v]     = (_Float16)s[K * stride + n0 + N];
    r[2 * v + 1] = (_Float16)s[(K + 1) * stride + n0 + N];
  }
  return r;
}

// Transposed B: element(K,N) = s[(n0+N)*stride + k0+K]  (K-pairs contiguous)
__device__ __forceinline__ v16h ldBT(const float* s, int stride, int k0, int n0) {
  const int lane = threadIdx.x & 31;
  const int N = lane & 15, half = lane >> 4;
  const float2* row2 = (const float2*)(s + (n0 + N) * stride + k0);
  v16h r;
#pragma unroll
  for (int v = 0; v < 8; ++v) {
    const int K = ((v & 3) << 1) + (half << 3) + ((v >> 2) << 4);
    float2 p = row2[K >> 1];
    r[2 * v]     = (_Float16)p.x;
    r[2 * v + 1] = (_Float16)p.y;
  }
  return r;
}

__device__ __forceinline__ v8f wmma_f16(v16h a, v16h b, v8f c) {
  return __builtin_amdgcn_wmma_f32_16x16x32_f16(false, a, false, b, (short)0, c,
                                                false, false);
}

__device__ __forceinline__ float clip60(float x) {
  return fminf(fmaxf(x, -60.f), 50.f);
}

// ---------------------------------------------------------------------------

__global__ __launch_bounds__(128) void gla_chunk_scan(
    const float* __restrict__ x, const float* __restrict__ q,
    const float* __restrict__ k, const float* __restrict__ v,
    const float* __restrict__ Wdw, const float* __restrict__ Wdb,
    float* __restrict__ out) {
  const int tid = threadIdx.x;
  const int wv  = tid >> 5;                 // wave 0..3
  const int blk = blockIdx.x;
  const int es  = blk & (ESPL - 1);
  const int h   = (blk >> 2) & (H - 1);
  const int b   = blk >> 5;
  const int e0  = es * ESL;

  __shared__ float Qs[C * QSTR];
  __shared__ float Ks[C * QSTR];
  __shared__ float Vs[C * VSTR];
  __shared__ float Asm[C * ASTR];
  __shared__ float Slds[Dh * SSTR];
  __shared__ float Ksum[Dh];
  __shared__ float abuf[C], rbuf[C], wexp[C], wst[C], dens[C];
  __shared__ float carryC;

  for (int i = tid; i < Dh * SSTR; i += 128) Slds[i] = 0.f;
  if (tid < Dh) Ksum[tid] = 0.f;
  if (tid == 0) carryC = 0.f;

  const float   bias = Wdb[h];
  const float4* Wr4  = (const float4*)(Wdw + (long)h * Dx);

  for (int ch = 0; ch < NC; ++ch) {
    const int t0 = ch * C;
    __syncthreads();  // protect LDS reuse across chunk iterations

    // ---- stage phi(q), phi(k) (32x64) and v e-slice (32x16) into LDS ----
#pragma unroll
    for (int m = 0; m < 4; ++m) {
      const int  vi = tid + 128 * m;        // float4 index 0..511
      const int  i  = vi >> 4, d4 = vi & 15;
      const long gi = ((long)((b * T + t0 + i) * H + h)) * (Dh / 4) + d4;
      float4 qq = ((const float4*)q)[gi];
      float4 kk = ((const float4*)k)[gi];
      float4 qf, kf;
      qf.x = fmaxf(qq.x, 0.f) + kEps; qf.y = fmaxf(qq.y, 0.f) + kEps;
      qf.z = fmaxf(qq.z, 0.f) + kEps; qf.w = fmaxf(qq.w, 0.f) + kEps;
      kf.x = fmaxf(kk.x, 0.f) + kEps; kf.y = fmaxf(kk.y, 0.f) + kEps;
      kf.z = fmaxf(kk.z, 0.f) + kEps; kf.w = fmaxf(kk.w, 0.f) + kEps;
      *(float4*)&Qs[i * QSTR + 4 * d4] = qf;   // 272B row base, 16B offsets
      *(float4*)&Ks[i * QSTR + 4 * d4] = kf;
    }
    {
      const int  i  = tid >> 2, d4 = tid & 3;                 // 32 rows x 4 f4
      const long gi = ((long)((b * T + t0 + i) * H + h)) * (Ev / 4) + (e0 >> 2) + d4;
      *(float4*)&Vs[i * VSTR + 4 * d4] = ((const float4*)v)[gi];
    }
    // ---- per-row decay logit a_i = -(relu(x_i . W_h + b_h) + eps) ----
    {
      const int i  = tid >> 2;      // row 0..31
      const int lg = tid & 3;       // 4 lanes per row (contiguous in wave)
      const float4* xr = (const float4*)(x + (long)(b * T + t0 + i) * Dx);
      float acc = 0.f;
#pragma unroll
      for (int m = 0; m < 32; ++m) {
        float4 xv  = xr[lg + 4 * m];
        float4 wv4 = Wr4[lg + 4 * m];
        acc += xv.x * wv4.x + xv.y * wv4.y + xv.z * wv4.z + xv.w * wv4.w;
      }
      acc += __shfl_xor(acc, 1, 32);
      acc += __shfl_xor(acc, 2, 32);
      if (lg == 0) abuf[i] = -(fmaxf(acc + bias, 0.f) + kEps);
    }
    __syncthreads();

    // ---- clipped cumulative log-decay, relative to chunk start ----
    if (tid == 0) {
      float Cr  = carryC;
      float cg0 = clip60(Cr);
      for (int i = 0; i < C; ++i) {
        Cr += abuf[i];
        rbuf[i] = clip60(Cr) - cg0;
      }
      carryC = Cr;
    }
    __syncthreads();
    if (tid < C) wexp[tid] = __expf(rbuf[tid]);
    if (tid >= 32 && tid < 64) {
      const int j = tid - 32;
      wst[j] = __expf(rbuf[C - 1] - rbuf[j]);
    }

    // ---- Phase A: A = causal_mask( (Q K^T) * exp(r_i - r_j) ) ----
    {
      const int ti = wv >> 1, tj = wv & 1;
      v8f acc = {};
      acc = wmma_f16(ldA(Qs, QSTR, 16 * ti, 0),  ldBT(Ks, QSTR, 0, 16 * tj),  acc);
      acc = wmma_f16(ldA(Qs, QSTR, 16 * ti, 32), ldBT(Ks, QSTR, 32, 16 * tj), acc);
      const int lane = tid & 31, half = lane >> 4, n = lane & 15;
      const int j    = 16 * tj + n;
      const int ib   = 16 * ti + 8 * half;          // 8 consecutive rows ib..ib+7
      // batch-load the 8 rbuf values + the single rbuf[j]; branch-free exp+mask
      const float4 rA = *(const float4*)&rbuf[ib];
      const float4 rB = *(const float4*)&rbuf[ib + 4];
      const float  rj = rbuf[j];
      float ri[8] = {rA.x, rA.y, rA.z, rA.w, rB.x, rB.y, rB.z, rB.w};
#pragma unroll
      for (int r = 0; r < 8; ++r) {
        const int   i = ib + r;
        const float e = __expf(ri[r] - rj);         // unconditional v_exp_f32
        const float f = (j <= i) ? e : 0.f;         // v_cndmask, no exec branch
        Asm[i * ASTR + j] = acc[r] * f;
      }
    }
    __syncthreads();

    // ---- Phase B: out = A V + (e^r * Q) S (waves 0,1); den (waves 2,3) ----
    v8f oacc = {};
    if (wv < 2) {
      const int ti = wv;
      oacc = wmma_f16(ldA(Asm, ASTR, 16 * ti, 0),                ldB(Vs, VSTR, 0, 0),    oacc);
      oacc = wmma_f16(ldA_rowscale(Qs, QSTR, 16 * ti, 0,  wexp), ldB(Slds, SSTR, 0, 0),  oacc);
      oacc = wmma_f16(ldA_rowscale(Qs, QSTR, 16 * ti, 32, wexp), ldB(Slds, SSTR, 32, 0), oacc);
    } else {
      // waves 2,3: den_i = 1e-5 + rowsum(Asm_i) + exp(r_i) * (q_i . Ksum)
      const int i = ((wv - 2) << 4) + (tid & 15);   // 32 rows over 64 threads
      if ((tid & 31) < 16) {
        const float4* ar = (const float4*)&Asm[i * ASTR];
        float s = 1e-5f;
#pragma unroll
        for (int j4 = 0; j4 < C / 4; ++j4) {
          float4 a4 = ar[j4];
          s += a4.x + a4.y + a4.z + a4.w;
        }
        const float4* qr = (const float4*)&Qs[i * QSTR];
        const float4* ks = (const float4*)Ksum;
        float dqk = 0.f;
#pragma unroll
        for (int d4 = 0; d4 < Dh / 4; ++d4) {
          float4 qv = qr[d4], kv = ks[d4];
          dqk += qv.x * kv.x + qv.y * kv.y + qv.z * kv.z + qv.w * kv.w;
        }
        dens[i] = s + wexp[i] * dqk;
      }
    }
    __syncthreads();

    // ---- normalize + store ----
    if (wv < 2) {
      const int ti = wv, lane = tid & 31, half = lane >> 4, n = lane & 15;
      const int ib = 16 * ti + 8 * half;
      const float4 dA = *(const float4*)&dens[ib];
      const float4 dB = *(const float4*)&dens[ib + 4];
      float dd[8] = {dA.x, dA.y, dA.z, dA.w, dB.x, dB.y, dB.z, dB.w};
#pragma unroll
      for (int r = 0; r < 8; ++r) {
        const int  i  = ib + r;
        const long go = ((long)((b * T + t0 + i) * H + h)) * Ev + e0 + n;
        out[go] = oacc[r] / dd[r];
      }
    }

    // ---- Phase C: S = e^{r_end} S + (w * K)^T V ; Ksum update ----
    {
      const float gend = wexp[C - 1];
      const int   m0   = 16 * wv;
      v8f sacc = {};
      sacc = wmma_f16(ldAT_kscale(Ks, QSTR, m0, 0, wst), ldB(Vs, VSTR, 0, 0), sacc);
      const int lane = tid & 31, half = lane >> 4, n = lane & 15;
#pragma unroll
      for (int r = 0; r < 8; ++r) {
        const int m = m0 + r + 8 * half;
        Slds[m * SSTR + n] = gend * Slds[m * SSTR + n] + sacc[r];
      }
      if (tid < Dh) {
        float a2 = gend * Ksum[tid];
#pragma unroll 4
        for (int j = 0; j < C; ++j) a2 += wst[j] * Ks[j * QSTR + tid];
        Ksum[tid] = a2;
      }
    }
  }
}

extern "C" void kernel_launch(void* const* d_in, const int* in_sizes, int n_in,
                              void* d_out, int out_size, void* d_ws, size_t ws_size,
                              hipStream_t stream) {
  (void)in_sizes; (void)n_in; (void)d_ws; (void)ws_size; (void)out_size;
  const float* x  = (const float*)d_in[0];
  const float* q  = (const float*)d_in[1];
  const float* k  = (const float*)d_in[2];
  const float* v  = (const float*)d_in[3];
  const float* Ww = (const float*)d_in[4];
  const float* Wb = (const float*)d_in[5];
  float* out = (float*)d_out;

  dim3 grid(Bb * H * ESPL);   // 64 workgroups
  dim3 block(128);            // 4 waves (wave32)
  hipLaunchKernelGGL(gla_chunk_scan, grid, block, 0, stream,
                     x, q, k, v, Ww, Wb, out);
}
